// PANNAcceptor_23742579212667
// MI455X (gfx1250) — matrix operand order repaired and checked by reading, not compiled, not run on hardware
//
#include <hip/hip_runtime.h>

// PANN acceptor: h <- W[x_t] @ h, 256x256 fp32, sequential over t, batch=128.
// Strategy: one 256-thread block (8 wave32) per batch element.
// Each step is done as WMMA bf16x3 (hi/lo split of W and h, drop lo*lo):
//   D(16x16) = A(16x32 of W) x B(32x16 of h replicated across columns)
// accumulated in f32 over 8 K-chunks. All 16 columns of D are identical;
// lanes 0/16 of each wave extract the 16x2 output rows and write h back to LDS.

#define A_SZ 100
#define N_ST 256
#define B_SZ 128
#define S_LEN 1024

typedef __attribute__((ext_vector_type(8)))  __bf16 v8bf;
typedef __attribute__((ext_vector_type(16))) __bf16 v16bf;
typedef __attribute__((ext_vector_type(8)))  float  v8f;

__device__ __forceinline__ v16bf bf_cat(v8bf a, v8bf b) {
  return __builtin_shufflevector(a, b, 0,1,2,3,4,5,6,7,8,9,10,11,12,13,14,15);
}

// Split fp32 W into bf16 hi + bf16 residual lo (Whi + Wlo ~= W to ~2^-16 rel).
__global__ void __launch_bounds__(256) split_w_kernel(const float* __restrict__ W,
                                                      __bf16* __restrict__ Whi,
                                                      __bf16* __restrict__ Wlo,
                                                      int n) {
  int i = blockIdx.x * blockDim.x + threadIdx.x;
  if (i >= n) return;
  float w  = W[i];
  __bf16 h = (__bf16)w;          // RNE
  Whi[i] = h;
  Wlo[i] = (__bf16)(w - (float)h);
}

__global__ void __launch_bounds__(256) pann_scan_kernel(
    const int*   __restrict__ xs,       // [B,S]
    const int*   __restrict__ lengths,  // [B]
    const __bf16* __restrict__ Whi,     // [A,N,N]
    const __bf16* __restrict__ Wlo,     // [A,N,N]
    const float* __restrict__ lin_w,    // [2,N]
    const float* __restrict__ lin_b,    // [2]
    float*       __restrict__ out) {    // [B,2]
  __shared__ __bf16 h_hi[2][N_ST];
  __shared__ __bf16 h_lo[2][N_ST];
  __shared__ float  h32 [2][N_ST];
  __shared__ float  red [N_ST];

  const int tid  = threadIdx.x;
  const int lane = tid & 31;
  const int wv   = tid >> 5;            // wave id 0..7
  const int b    = blockIdx.x;
  const int L    = lengths[b];
  const int* xrow = xs + b * S_LEN;

  // h0 = one-hot(state 0)
  h32 [0][tid] = (tid == 0) ? 1.0f : 0.0f;
  h_hi[0][tid] = (__bf16)((tid == 0) ? 1.0f : 0.0f);
  h_lo[0][tid] = (__bf16)0.0f;
  __syncthreads();

  const int laneHi = lane >> 4;         // 0 | 1 (lane half)
  const int i0     = wv * 16;           // tile0 rows [i0, i0+16); tile1 = +128
  const int row0   = i0 + (lane & 15);  // A row this lane loads
  const int row1   = row0 + 128;
  const int colb   = laneHi * 8;        // A K-offset: K0..7 (lo lanes) / K8..15 (hi lanes)

  int cur = 0;
  for (int t = 0; t < L; ++t) {
    const int x = __builtin_amdgcn_readfirstlane(xrow[t]);   // uniform symbol -> SGPR base
    const __bf16* Wh = Whi + (size_t)x * (N_ST * N_ST);
    const __bf16* Wl = Wlo + (size_t)x * (N_ST * N_ST);
    v8f d0 = {};   // rows [i0, i0+16)
    v8f d1 = {};   // rows [i0+128, i0+144)
#pragma unroll
    for (int c = 0; c < 8; ++c) {       // K chunks of 32
      // B fragment = h[c*32 + laneHi*8 + {0..7, 16..23}] — replicated over N
      const int kb = c * 32 + laneHi * 8;
      const v8bf* hp = (const v8bf*)&h_hi[cur][kb];
      const v8bf* lp = (const v8bf*)&h_lo[cur][kb];
      v16bf bh = bf_cat(hp[0], hp[2]);
      v16bf bl = bf_cat(lp[0], lp[2]);
      // A fragments: two b128 per matrix (K +0..7/+8..15 then +16..23/+24..31)
      const int cb = c * 32 + colb;
      const v8bf* a0h = (const v8bf*)(Wh + row0 * N_ST + cb);
      const v8bf* a0l = (const v8bf*)(Wl + row0 * N_ST + cb);
      const v8bf* a1h = (const v8bf*)(Wh + row1 * N_ST + cb);
      const v8bf* a1l = (const v8bf*)(Wl + row1 * N_ST + cb);
      v16bf A0h = bf_cat(a0h[0], a0h[2]);
      v16bf A0l = bf_cat(a0l[0], a0l[2]);
      v16bf A1h = bf_cat(a1h[0], a1h[2]);
      v16bf A1l = bf_cat(a1l[0], a1l[2]);
      // bf16x3: hi*hi + hi*lo + lo*hi  (lo*lo ~ 2^-32, dropped)
      d0 = __builtin_amdgcn_wmma_f32_16x16x32_bf16(false, A0h, false, bh, (short)0, d0, false, false);
      d0 = __builtin_amdgcn_wmma_f32_16x16x32_bf16(false, A0h, false, bl, (short)0, d0, false, false);
      d0 = __builtin_amdgcn_wmma_f32_16x16x32_bf16(false, A0l, false, bh, (short)0, d0, false, false);
      d1 = __builtin_amdgcn_wmma_f32_16x16x32_bf16(false, A1h, false, bh, (short)0, d1, false, false);
      d1 = __builtin_amdgcn_wmma_f32_16x16x32_bf16(false, A1h, false, bl, (short)0, d1, false, false);
      d1 = __builtin_amdgcn_wmma_f32_16x16x32_bf16(false, A1l, false, bh, (short)0, d1, false, false);
    }
    // Extract h_new: VGPR r of lane 0 -> M=r; of lane 16 -> M=8+r (all N equal).
    const int nxt = cur ^ 1;
    if ((lane & 15) == 0) {
      const int mb = i0 + laneHi * 8;
#pragma unroll
      for (int r = 0; r < 8; ++r) {
        float f = d0[r];
        __bf16 fh = (__bf16)f;
        h32 [nxt][mb + r] = f;
        h_hi[nxt][mb + r] = fh;
        h_lo[nxt][mb + r] = (__bf16)(f - (float)fh);
        float g = d1[r];
        __bf16 gh = (__bf16)g;
        h32 [nxt][mb + 128 + r] = g;
        h_hi[nxt][mb + 128 + r] = gh;
        h_lo[nxt][mb + 128 + r] = (__bf16)(g - (float)gh);
      }
    }
    __syncthreads();
    cur = nxt;
  }

  // Head: out[b][o] = dot(lin_w[o], h_final) + lin_b[o]
  const float hv = h32[cur][tid];
#pragma unroll
  for (int o = 0; o < 2; ++o) {
    red[tid] = hv * lin_w[o * N_ST + tid];
    __syncthreads();
    for (int s = 128; s > 0; s >>= 1) {
      if (tid < s) red[tid] += red[tid + s];
      __syncthreads();
    }
    if (tid == 0) out[2 * b + o] = red[0] + lin_b[o];
    __syncthreads();
  }
}

extern "C" void kernel_launch(void* const* d_in, const int* in_sizes, int n_in,
                              void* d_out, int out_size, void* d_ws, size_t ws_size,
                              hipStream_t stream) {
  const int*   xs      = (const int*)d_in[0];
  const int*   lengths = (const int*)d_in[1];
  const float* W       = (const float*)d_in[2];
  const float* lin_w   = (const float*)d_in[3];
  const float* lin_b   = (const float*)d_in[4];
  float*       out     = (float*)d_out;

  const size_t nW = (size_t)A_SZ * N_ST * N_ST;   // 6.55M elems; hi+lo = 26.2 MB in ws
  __bf16* Whi = (__bf16*)d_ws;
  __bf16* Wlo = Whi + nW;

  split_w_kernel<<<(int)((nW + 255) / 256), 256, 0, stream>>>(W, Whi, Wlo, (int)nW);
  pann_scan_kernel<<<B_SZ, 256, 0, stream>>>(xs, lengths, Whi, Wlo, lin_w, lin_b, out);
}
